// MutPredV2Model_9088150798462
// MI455X (gfx1250) — matrix operand order repaired and checked by reading, not compiled, not run on hardware
//
#include <hip/hip_runtime.h>
#include <math.h>

typedef __bf16 bf16;
typedef __attribute__((ext_vector_type(16))) bf16 v16bf;
typedef __attribute__((ext_vector_type(8)))  bf16 bf16x8;
typedef __attribute__((ext_vector_type(8)))  float v8f;

// LDS strides padded to 40 elems (80B): ds_load_b128 phase banks = n*20 mod 64,
// 16 distinct multiples of 4 -> conflict-free (stride 32 would be 4-way conflicted).
#define LDS_STRIDE 40

union AFrag { v16bf v; bf16 e[16]; struct { bf16x8 lo; bf16x8 hi; } h; };

__device__ __forceinline__ v8f wmma_bf16f32(v16bf a, v16bf b, v8f c) {
  return __builtin_amdgcn_wmma_f32_16x16x32_bf16(false, a, false, b, (short)0, c, false, false);
}

// CDNA5 async copy: global -> LDS, 16B per lane, tracked by ASYNCcnt.
__device__ __forceinline__ void async_copy_b128(unsigned lds_off, const void* gaddr) {
  asm volatile("global_load_async_to_lds_b128 %0, %1, off"
               :: "v"(lds_off), "v"(gaddr) : "memory");
}

// CDNA5 packed bf16 f-atomic (no return): 2 adds per atomic op.
__device__ __forceinline__ void atomic_pk_add_bf16(bf16* addr, float a, float b) {
  union { bf16 h[2]; unsigned u; } p;
  p.h[0] = (bf16)a; p.h[1] = (bf16)b;
  asm volatile("global_atomic_pk_add_bf16 %0, %1, off"
               :: "v"((unsigned long long)(uintptr_t)addr), "v"(p.u) : "memory");
}

// A-matrix 16x32 bf16 fragment (ISA 7.12.2): lane holds row m=lane&15.
// half==0: K 0..7,16..23 ; half==1: K 8..15,24..31
__device__ __forceinline__ v16bf load_a_global(const bf16* __restrict__ row, int kc, int half) {
  AFrag u;
  u.h.lo = *(const bf16x8*)(row + kc * 32 + half * 8);
  u.h.hi = *(const bf16x8*)(row + kc * 32 + 16 + half * 8);
  return u.v;
}

__device__ __forceinline__ v16bf load_a_lds(const bf16* base, int mrow, int stride, int half) {
  const bf16* row = base + mrow * stride;
  AFrag u;
  u.h.lo = *(const bf16x8*)(row + half * 8);
  u.h.hi = *(const bf16x8*)(row + 16 + half * 8);
  return u.v;
}

// B chunk staged in LDS as [256 cols][LDS_STRIDE] (32 K-values used): half h holds K 8h..8h+7, 16+8h..23+8h.
__device__ __forceinline__ v16bf load_b_lds(const bf16* B, int n0, int lane) {
  int n = n0 + (lane & 15);
  int half = lane >> 4;
  AFrag u;
  u.h.lo = *(const bf16x8*)(B + n * LDS_STRIDE + half * 8);
  u.h.hi = *(const bf16x8*)(B + n * LDS_STRIDE + 16 + half * 8);
  return u.v;
}

// B fragment straight from packed global weights [K/32][N][32].
__device__ __forceinline__ v16bf load_b_global(const bf16* __restrict__ Wp, int N, int kc, int n0, int lane) {
  int n = n0 + (lane & 15);
  int half = lane >> 4;
  AFrag u;
  u.h.lo = *(const bf16x8*)(Wp + (((size_t)kc * N + n) << 5) + half * 8);
  u.h.hi = *(const bf16x8*)(Wp + (((size_t)kc * N + n) << 5) + 16 + half * 8);
  return u.v;
}

// 16 C-accumulators (16x256 f32) -> 8 bf16 A-fragments via per-wave LDS bounce; bias+relu.
__device__ __forceinline__ void acc_to_frags(const v8f* acc, const float* __restrict__ bias,
                                             bf16* stg, int mrow, int half, int lane, v16bf* out) {
  int col16 = lane & 15;
#pragma unroll
  for (int c = 0; c < 8; ++c) {
#pragma unroll
    for (int j = 0; j < 2; ++j) {
      float bv = bias ? bias[c * 32 + j * 16 + col16] : 0.f;
      const v8f a = acc[2 * c + j];
#pragma unroll
      for (int r = 0; r < 8; ++r) {
        stg[(half * 8 + r) * LDS_STRIDE + j * 16 + col16] = (bf16)fmaxf(a[r] + bv, 0.f);
      }
    }
    asm volatile("s_wait_dscnt 0x0" ::: "memory");
    out[c] = load_a_lds(stg, mrow, LDS_STRIDE, half);
    asm volatile("s_wait_dscnt 0x0" ::: "memory");
  }
}

// ---------------- weight packing / dtype conversion ----------------

__global__ void pack_w_kernel(const float* __restrict__ W, bf16* __restrict__ Wp,
                              int K, int Kpad, int N) {
  int i = blockIdx.x * blockDim.x + threadIdx.x;
  if (i >= Kpad * N) return;
  int k = i / N, n = i % N;
  float v = (k < K) ? W[(size_t)k * N + n] : 0.f;
  Wp[(((size_t)(k >> 5) * N + n) << 5) + (k & 31)] = (bf16)v;
}

__global__ void cvt_bf16_kernel(const float* __restrict__ X, bf16* __restrict__ Y, int total) {
  int i = blockIdx.x * blockDim.x + threadIdx.x;
  if (i < total) Y[i] = (bf16)X[i];
}

// ---------------- generic WMMA GEMM with split-A (concat) support ----------------

__global__ __launch_bounds__(256, 2) void gemm_kernel(
    const bf16* __restrict__ A, int lda,
    const bf16* __restrict__ A2, int lda2, int kc_split,
    const bf16* __restrict__ Wp, const float* __restrict__ bias,
    float* __restrict__ Cf, int ldcf, bf16* __restrict__ Cb, int ldcb,
    int M, int N, int K, int relu_flag) {
  __shared__ bf16 atile[128 * LDS_STRIDE];
  int tid = threadIdx.x, lane = tid & 31, wv = tid >> 5;
  int half = lane >> 4, mrow = lane & 15;
  int row0 = blockIdx.x * 128;
  int col0 = blockIdx.y * 64;
  v8f vz = {0.f, 0.f, 0.f, 0.f, 0.f, 0.f, 0.f, 0.f};
  v8f acc[4];
#pragma unroll
  for (int i = 0; i < 4; ++i) acc[i] = vz;

  int KC = K >> 5;
  for (int kc = 0; kc < KC; ++kc) {
    {  // async-stage A chunk [128 x 32] bf16 (padded rows): 2 threads/row, 32B each
      int r = tid >> 1;
      int gr = row0 + r; if (gr > M - 1) gr = M - 1;
      const bf16* base = (kc < kc_split) ? (A + (size_t)gr * lda + kc * 32)
                                         : (A2 + (size_t)gr * lda2 + (kc - kc_split) * 32);
      const char* g = (const char*)base + (tid & 1) * 32;
      unsigned l = (unsigned)(uintptr_t)(atile + r * LDS_STRIDE) + (tid & 1) * 32;
      async_copy_b128(l, g);
      async_copy_b128(l + 16, g + 16);
    }
    asm volatile("s_wait_asynccnt 0x0" ::: "memory");
    __syncthreads();
    v16bf a = load_a_lds(atile, wv * 16 + mrow, LDS_STRIDE, half);
#pragma unroll
    for (int nb = 0; nb < 4; ++nb) {
      v16bf b = load_b_global(Wp, N, kc, col0 + nb * 16, lane);
      acc[nb] = wmma_bf16f32(a, b, acc[nb]);
    }
    __syncthreads();
  }
#pragma unroll
  for (int nb = 0; nb < 4; ++nb) {
    int col = col0 + nb * 16 + mrow;
    float bv = bias ? bias[col] : 0.f;
#pragma unroll
    for (int r = 0; r < 8; ++r) {
      int grow = row0 + wv * 16 + half * 8 + r;
      if (grow < M) {
        float v = acc[nb][r] + bv;
        if (relu_flag) v = fmaxf(v, 0.f);
        if (Cf) Cf[(size_t)grow * ldcf + col] = v;
        if (Cb) Cb[(size_t)grow * ldcb + col] = (bf16)v;
      }
    }
  }
}

// ---------------- fusion encoder tail ----------------

__global__ void gate_kernel(const float* __restrict__ g1, const float* __restrict__ w2,
                            const float* __restrict__ b2, float* __restrict__ gate, int M) {
  int w = blockIdx.x * (blockDim.x >> 5) + (threadIdx.x >> 5);
  int lane = threadIdx.x & 31;
  if (w >= M) return;
  const float* row = g1 + (size_t)w * 256;
  float s = 0.f;
#pragma unroll
  for (int j = 0; j < 8; ++j) { int i = lane + 32 * j; s += row[i] * w2[i]; }
  s += __shfl_xor(s, 1, 32); s += __shfl_xor(s, 2, 32);
  s += __shfl_xor(s, 4, 32); s += __shfl_xor(s, 8, 32); s += __shfl_xor(s, 16, 32);
  if (lane == 0) gate[w] = 1.f / (1.f + expf(-(s + b2[0])));
}

__global__ void fuse_kernel(const float* __restrict__ hs, const float* __restrict__ he,
                            const float* __restrict__ gate, const float* __restrict__ ss,
                            const float* __restrict__ es, float* __restrict__ h,
                            bf16* __restrict__ h_bf, int M) {
  int i = blockIdx.x * blockDim.x + threadIdx.x;
  if (i >= M * 256) return;
  int r = i >> 8;
  float g = gate[r];
  float v = fmaxf(g * ss[0] * hs[i] + (1.f - g) * es[0] * he[i], 0.f);
  h[i] = v;
  h_bf[i] = (bf16)v;
}

// ---------------- edge message kernel (dominant compute) ----------------

__global__ __launch_bounds__(256, 1) void edge_kernel(
    const bf16* __restrict__ hbf, const float* __restrict__ pos,
    const int* __restrict__ srcI, const int* __restrict__ dstI,
    const bf16* __restrict__ eattr,
    const bf16* __restrict__ w1p, const float* __restrict__ b1,
    const bf16* __restrict__ w2p, const float* __restrict__ b2,
    const bf16* __restrict__ cw1p, const float* __restrict__ cb1,
    const float* __restrict__ cw2, const float* __restrict__ cb2,
    bf16* __restrict__ agg, float* __restrict__ posn, int E) {
  __shared__ bf16 Bst[2][256 * LDS_STRIDE];  // double-buffered weight chunk (padded)
  __shared__ bf16 tail[8][16 * LDS_STRIDE];
  __shared__ bf16 stage[8][16 * LDS_STRIDE];
  __shared__ float cbuf[8][16];

  int tid = threadIdx.x, lane = tid & 31, wv = tid >> 5;
  int half = lane >> 4, mrow = lane & 15, col16 = lane & 15;

  // stage one 16KB weight chunk: thread t copies column t (64B) into padded LDS slot
  auto issueB = [&](const bf16* Wp, int kc, int buf) {
    const char* g = (const char*)(Wp + (size_t)kc * 8192) + tid * 64;
    unsigned l = (unsigned)(uintptr_t)(&Bst[buf][0]) + tid * (LDS_STRIDE * 2);
#pragma unroll
    for (int i = 0; i < 4; ++i) async_copy_b128(l + i * 16, g + i * 16);
  };

  int e = blockIdx.x * 128 + wv * 16 + mrow;
  if (e > E - 1) e = E - 1;
  int s = srcI[e], d = dstI[e];
  float dx = pos[d * 3 + 0] - pos[s * 3 + 0];
  float dy = pos[d * 3 + 1] - pos[s * 3 + 1];
  float dz = pos[d * 3 + 2] - pos[s * 3 + 2];
  float dist2 = fminf(dx * dx + dy * dy + dz * dz, 1000.f);
  float inv = rsqrtf(dist2 + 1e-8f);

  issueB(w1p, 0, 0);  // prefetch first weight chunk

  if (half == 0) {  // tail chunk of ei: [dist2 | edge_attr(16) | zeros(15)]
    bf16* t = &tail[wv][mrow * LDS_STRIDE];
    t[0] = (bf16)dist2;
    const bf16* ea = eattr + (size_t)e * 16;
#pragma unroll
    for (int j = 0; j < 16; ++j) t[1 + j] = ea[j];
#pragma unroll
    for (int j = 17; j < 32; ++j) t[j] = (bf16)0.f;
  }
  asm volatile("s_wait_dscnt 0x0" ::: "memory");

  const bf16* hs_row = hbf + (size_t)s * 256;
  const bf16* hd_row = hbf + (size_t)d * 256;
  v8f vz = {0.f, 0.f, 0.f, 0.f, 0.f, 0.f, 0.f, 0.f};
  v8f acc[16];
#pragma unroll
  for (int i = 0; i < 16; ++i) acc[i] = vz;

  // GEMM1: m1 = relu(ei @ W1 + b1), K = 544 (17 chunks)
  for (int kc = 0; kc < 17; ++kc) {
    if (kc + 1 < 17) {
      issueB(w1p, kc + 1, (kc + 1) & 1);
      asm volatile("s_wait_asynccnt 0x4" ::: "memory");
    } else {
      asm volatile("s_wait_asynccnt 0x0" ::: "memory");
    }
    __syncthreads();
    v16bf a;
    if (kc < 8)       a = load_a_global(hs_row, kc, half);
    else if (kc < 16) a = load_a_global(hd_row, kc - 8, half);
    else              a = load_a_lds(&tail[wv][0], mrow, LDS_STRIDE, half);
    const bf16* bb = &Bst[kc & 1][0];
#pragma unroll
    for (int nb = 0; nb < 16; ++nb) {
      v16bf b = load_b_lds(bb, nb * 16, lane);
      acc[nb] = wmma_bf16f32(a, b, acc[nb]);
    }
    __syncthreads();
  }
  issueB(w2p, 0, 0);
  v16bf m1f[8];
  acc_to_frags(acc, b1, &stage[wv][0], mrow, half, lane, m1f);

  // GEMM2: m = relu(m1 @ W2 + b2), K = 256
#pragma unroll
  for (int i = 0; i < 16; ++i) acc[i] = vz;
#pragma unroll
  for (int kc = 0; kc < 8; ++kc) {
    if (kc + 1 < 8) {
      issueB(w2p, kc + 1, (kc + 1) & 1);
      asm volatile("s_wait_asynccnt 0x4" ::: "memory");
    } else {
      asm volatile("s_wait_asynccnt 0x0" ::: "memory");
    }
    __syncthreads();
    const bf16* bb = &Bst[kc & 1][0];
#pragma unroll
    for (int nb = 0; nb < 16; ++nb) {
      v16bf b = load_b_lds(bb, nb * 16, lane);
      acc[nb] = wmma_bf16f32(m1f[kc], b, acc[nb]);
    }
    __syncthreads();
  }
  issueB(cw1p, 0, 0);
  v16bf m2f[8];
  acc_to_frags(acc, b2, &stage[wv][0], mrow, half, lane, m2f);

  // GEMM3 (coord_w1) fused with coord_w2 dot -> scalar c per edge
#pragma unroll
  for (int i = 0; i < 16; ++i) acc[i] = vz;
#pragma unroll
  for (int kc = 0; kc < 8; ++kc) {
    if (kc + 1 < 8) {
      issueB(cw1p, kc + 1, (kc + 1) & 1);
      asm volatile("s_wait_asynccnt 0x4" ::: "memory");
    } else {
      asm volatile("s_wait_asynccnt 0x0" ::: "memory");
    }
    __syncthreads();
    const bf16* bb = &Bst[kc & 1][0];
#pragma unroll
    for (int nb = 0; nb < 16; ++nb) {
      v16bf b = load_b_lds(bb, nb * 16, lane);
      acc[nb] = wmma_bf16f32(m2f[kc], b, acc[nb]);
    }
    __syncthreads();
  }
  float cpart[8];
#pragma unroll
  for (int r = 0; r < 8; ++r) cpart[r] = 0.f;
#pragma unroll
  for (int nb = 0; nb < 16; ++nb) {
    int col = nb * 16 + col16;
    float bv = cb1[col], wv2 = cw2[col];
#pragma unroll
    for (int r = 0; r < 8; ++r) cpart[r] += fmaxf(acc[nb][r] + bv, 0.f) * wv2;
  }
#pragma unroll
  for (int r = 0; r < 8; ++r) {
    float sv = cpart[r];
    sv += __shfl_xor(sv, 1, 32); sv += __shfl_xor(sv, 2, 32);
    sv += __shfl_xor(sv, 4, 32); sv += __shfl_xor(sv, 8, 32);
    cpart[r] = sv;
  }
  if (col16 == 0) {
#pragma unroll
    for (int r = 0; r < 8; ++r)
      cbuf[wv][half * 8 + r] = tanhf(cpart[r] + cb2[0]) * 0.1f;
  }
  asm volatile("s_wait_dscnt 0x0" ::: "memory");

  if (half == 0) {  // position scatter (one lane per edge)
    float cv = cbuf[wv][mrow];
    atomicAdd(&posn[d * 3 + 0], dx * inv * cv);
    atomicAdd(&posn[d * 3 + 1], dy * inv * cv);
    atomicAdd(&posn[d * 3 + 2], dz * inv * cv);
  }

  // agg scatter: packed bf16 atomics, 2 columns per op
  bf16* arow = agg + (size_t)d * 256;
#pragma unroll
  for (int kc = 0; kc < 8; ++kc) {
    AFrag u; u.v = m2f[kc];
#pragma unroll
    for (int i = 0; i < 8; i += 2) {
      int colA = kc * 32 + half * 8 + i;         // elems i, i+1
      atomic_pk_add_bf16(arow + colA, (float)u.e[i], (float)u.e[i + 1]);
      int colB = kc * 32 + 16 + half * 8 + i;    // elems i+8, i+9
      atomic_pk_add_bf16(arow + colB, (float)u.e[i + 8], (float)u.e[i + 9]);
    }
  }
}

// ---------------- layernorm (wave per row) + head ----------------

__global__ void ln_kernel(const float* __restrict__ x, const float* __restrict__ resid,
                          const float* __restrict__ g, const float* __restrict__ b,
                          float* __restrict__ out, bf16* __restrict__ outb, int M) {
  int w = blockIdx.x * (blockDim.x >> 5) + (threadIdx.x >> 5);
  int lane = threadIdx.x & 31;
  if (w >= M) return;
  const float* xr = x + (size_t)w * 256;
  const float* rr = resid ? resid + (size_t)w * 256 : nullptr;
  float v[8], s = 0.f;
#pragma unroll
  for (int j = 0; j < 8; ++j) {
    int c = lane + 32 * j;
    v[j] = xr[c] + (rr ? rr[c] : 0.f);
    s += v[j];
  }
  s += __shfl_xor(s, 1, 32); s += __shfl_xor(s, 2, 32);
  s += __shfl_xor(s, 4, 32); s += __shfl_xor(s, 8, 32); s += __shfl_xor(s, 16, 32);
  float mu = s * (1.f / 256.f);
  float var = 0.f;
#pragma unroll
  for (int j = 0; j < 8; ++j) { float dd = v[j] - mu; var += dd * dd; }
  var += __shfl_xor(var, 1, 32); var += __shfl_xor(var, 2, 32);
  var += __shfl_xor(var, 4, 32); var += __shfl_xor(var, 8, 32); var += __shfl_xor(var, 16, 32);
  float rs = rsqrtf(var * (1.f / 256.f) + 1e-5f);
#pragma unroll
  for (int j = 0; j < 8; ++j) {
    int c = lane + 32 * j;
    float o = (v[j] - mu) * rs * g[c] + b[c];
    out[(size_t)w * 256 + c] = o;
    if (outb) outb[(size_t)w * 256 + c] = (bf16)o;
  }
}

__global__ void head_kernel(const float* __restrict__ h, const float* __restrict__ W,
                            const float* __restrict__ b, float* __restrict__ out, int M) {
  int i = blockIdx.x * blockDim.x + threadIdx.x;
  if (i >= M * 20) return;
  int r = i / 20, c = i % 20;
  float s = b[c];
  const float* hr = h + (size_t)r * 256;
  for (int k = 0; k < 256; ++k) s += hr[k] * W[k * 20 + c];
  out[i] = s;
}

// ---------------- host orchestration ----------------

extern "C" void kernel_launch(void* const* d_in, const int* in_sizes, int n_in,
                              void* d_out, int out_size, void* d_ws, size_t ws_size,
                              hipStream_t stream) {
  const int N = 10000, E = 320000, L = 4;
  (void)in_sizes; (void)n_in; (void)out_size; (void)ws_size;

  const float* x_struct = (const float*)d_in[0];
  const float* x_esm    = (const float*)d_in[1];
  const int*   eidx     = (const int*)d_in[2];
  const float* edge_attr= (const float*)d_in[3];
  const float* pos      = (const float*)d_in[4];
  const float* struct_w = (const float*)d_in[5];
  const float* struct_b = (const float*)d_in[6];
  const float* esm_w    = (const float*)d_in[7];
  const float* esm_b    = (const float*)d_in[8];
  const float* gate_w1  = (const float*)d_in[9];
  const float* gate_b1  = (const float*)d_in[10];
  const float* gate_w2  = (const float*)d_in[11];
  const float* gate_b2  = (const float*)d_in[12];
  const float* s_scale  = (const float*)d_in[13];
  const float* e_scale  = (const float*)d_in[14];
  const float* edge_w1  = (const float*)d_in[15];
  const float* edge_b1  = (const float*)d_in[16];
  const float* edge_w2  = (const float*)d_in[17];
  const float* edge_b2  = (const float*)d_in[18];
  const float* coord_w1 = (const float*)d_in[19];
  const float* coord_b1 = (const float*)d_in[20];
  const float* coord_w2 = (const float*)d_in[21];
  const float* coord_b2 = (const float*)d_in[22];
  const float* node_w1  = (const float*)d_in[23];
  const float* node_b1  = (const float*)d_in[24];
  const float* ln_g     = (const float*)d_in[25];
  const float* ln_b     = (const float*)d_in[26];
  const float* fn_g     = (const float*)d_in[27];
  const float* fn_b     = (const float*)d_in[28];
  const float* head_w   = (const float*)d_in[29];
  const float* head_b   = (const float*)d_in[30];
  const int* srcI = eidx;
  const int* dstI = eidx + E;

  char* ws = (char*)d_ws;
  size_t off = 0;
  auto alloc = [&](size_t bytes) -> void* {
    off = (off + 255) & ~(size_t)255;
    void* p = ws + off; off += bytes; return p;
  };

  bf16* struct_wp = (bf16*)alloc((size_t)64   * 256 * sizeof(bf16));
  bf16* esm_wp    = (bf16*)alloc((size_t)1280 * 256 * sizeof(bf16));
  bf16* gate_w1p  = (bf16*)alloc((size_t)512  * 256 * sizeof(bf16));
  size_t ew1_sz = (size_t)544 * 256, ew2_sz = (size_t)256 * 256, nw1_sz = (size_t)512 * 256;
  bf16* edge_w1p  = (bf16*)alloc(L * ew1_sz * sizeof(bf16));
  bf16* edge_w2p  = (bf16*)alloc(L * ew2_sz * sizeof(bf16));
  bf16* coord_w1p = (bf16*)alloc(L * ew2_sz * sizeof(bf16));
  bf16* node_w1p  = (bf16*)alloc(L * nw1_sz * sizeof(bf16));
  bf16* xs_bf  = (bf16*)alloc((size_t)N * 64   * sizeof(bf16));
  bf16* xe_bf  = (bf16*)alloc((size_t)N * 1280 * sizeof(bf16));
  bf16* ea_bf  = (bf16*)alloc((size_t)E * 16   * sizeof(bf16));
  bf16* gi_bf  = (bf16*)alloc((size_t)N * 512  * sizeof(bf16));
  bf16* h_bf   = (bf16*)alloc((size_t)N * 256  * sizeof(bf16));
  bf16* agg_bf = (bf16*)alloc((size_t)N * 256  * sizeof(bf16));
  float* hs_f  = (float*)alloc((size_t)N * 256 * 4);
  float* he_f  = (float*)alloc((size_t)N * 256 * 4);
  float* g1_f  = (float*)alloc((size_t)N * 256 * 4);
  float* gate  = (float*)alloc((size_t)N * 4);
  float* h_f   = (float*)alloc((size_t)N * 256 * 4);
  float* hu_f  = (float*)alloc((size_t)N * 256 * 4);
  float* hfin  = (float*)alloc((size_t)N * 256 * 4);
  float* posA  = (float*)alloc((size_t)N * 3 * 4);
  float* posB  = (float*)alloc((size_t)N * 3 * 4);

  auto cdiv = [](int a, int b) { return (a + b - 1) / b; };

  pack_w_kernel<<<cdiv(64 * 256, 256), 256, 0, stream>>>(struct_w, struct_wp, 64, 64, 256);
  pack_w_kernel<<<cdiv(1280 * 256, 256), 256, 0, stream>>>(esm_w, esm_wp, 1280, 1280, 256);
  pack_w_kernel<<<cdiv(512 * 256, 256), 256, 0, stream>>>(gate_w1, gate_w1p, 512, 512, 256);
  for (int l = 0; l < L; ++l) {
    pack_w_kernel<<<cdiv(544 * 256, 256), 256, 0, stream>>>(edge_w1 + (size_t)l * 529 * 256, edge_w1p + l * ew1_sz, 529, 544, 256);
    pack_w_kernel<<<cdiv(256 * 256, 256), 256, 0, stream>>>(edge_w2 + (size_t)l * 256 * 256, edge_w2p + l * ew2_sz, 256, 256, 256);
    pack_w_kernel<<<cdiv(256 * 256, 256), 256, 0, stream>>>(coord_w1 + (size_t)l * 256 * 256, coord_w1p + l * ew2_sz, 256, 256, 256);
    pack_w_kernel<<<cdiv(512 * 256, 256), 256, 0, stream>>>(node_w1 + (size_t)l * 512 * 256, node_w1p + l * nw1_sz, 512, 512, 256);
  }

  cvt_bf16_kernel<<<cdiv(N * 64, 256), 256, 0, stream>>>(x_struct, xs_bf, N * 64);
  cvt_bf16_kernel<<<cdiv(N * 1280, 256), 256, 0, stream>>>(x_esm, xe_bf, N * 1280);
  cvt_bf16_kernel<<<cdiv(E * 16, 256), 256, 0, stream>>>(edge_attr, ea_bf, E * 16);

  // fusion encoder
  dim3 gN(cdiv(N, 128), 4);
  gemm_kernel<<<gN, 256, 0, stream>>>(xs_bf, 64, nullptr, 0, 1 << 20, struct_wp, struct_b,
                                      hs_f, 256, gi_bf, 512, N, 256, 64, 0);
  gemm_kernel<<<gN, 256, 0, stream>>>(xe_bf, 1280, nullptr, 0, 1 << 20, esm_wp, esm_b,
                                      he_f, 256, gi_bf + 256, 512, N, 256, 1280, 0);
  gemm_kernel<<<gN, 256, 0, stream>>>(gi_bf, 512, nullptr, 0, 1 << 20, gate_w1p, gate_b1,
                                      g1_f, 256, nullptr, 0, N, 256, 512, 1);
  gate_kernel<<<cdiv(N, 8), 256, 0, stream>>>(g1_f, gate_w2, gate_b2, gate, N);
  fuse_kernel<<<cdiv(N * 256, 256), 256, 0, stream>>>(hs_f, he_f, gate, s_scale, e_scale, h_f, h_bf, N);
  hipMemcpyAsync(posA, pos, (size_t)N * 3 * 4, hipMemcpyDeviceToDevice, stream);

  // EGNN layers
  float* pa = posA; float* pb = posB;
  for (int l = 0; l < L; ++l) {
    hipMemsetAsync(agg_bf, 0, (size_t)N * 256 * sizeof(bf16), stream);
    hipMemcpyAsync(pb, pa, (size_t)N * 3 * 4, hipMemcpyDeviceToDevice, stream);
    edge_kernel<<<E / 128, 256, 0, stream>>>(
        h_bf, pa, srcI, dstI, ea_bf,
        edge_w1p + l * ew1_sz, edge_b1 + l * 256,
        edge_w2p + l * ew2_sz, edge_b2 + l * 256,
        coord_w1p + l * ew2_sz, coord_b1 + l * 256,
        coord_w2 + l * 256, coord_b2 + l,
        agg_bf, pb, E);
    gemm_kernel<<<gN, 256, 0, stream>>>(h_bf, 256, agg_bf, 256, 8,
                                        node_w1p + l * nw1_sz, node_b1 + l * 256,
                                        hu_f, 256, nullptr, 0, N, 256, 512, 1);
    ln_kernel<<<cdiv(N, 8), 256, 0, stream>>>(h_f, hu_f, ln_g + l * 256, ln_b + l * 256,
                                              h_f, h_bf, N);
    float* t = pa; pa = pb; pb = t;
  }

  ln_kernel<<<cdiv(N, 8), 256, 0, stream>>>(h_f, nullptr, fn_g, fn_b, hfin, nullptr, N);
  head_kernel<<<cdiv(N * 20, 256), 256, 0, stream>>>(hfin, head_w, head_b, (float*)d_out, N);
}
